// PostProcessor_5892695130359
// MI455X (gfx1250) — compile-verified, matrix operand order checked by reading, not run on hardware
//
#include <hip/hip_runtime.h>
#include <hip/hip_bf16.h>

typedef __attribute__((ext_vector_type(16))) _Float16 v16h;
typedef __attribute__((ext_vector_type(8)))  float    v8f;

typedef __attribute__((address_space(1))) int as1_int;
typedef __attribute__((address_space(3))) int as3_int;

#if defined(__has_builtin)
#if __has_builtin(__builtin_amdgcn_global_load_async_to_lds_b32)
#define HAVE_ASYNC_LDS 1
#endif
#if __has_builtin(__builtin_amdgcn_s_wait_asynccnt)
#define HAVE_WAIT_ASYNC 1
#endif
#endif

#define N_PROP    4096
#define DETS      100
#define SCORE_TH  0.05f
#define NMS_TH    0.5f
#define IMG_WM1   511.0f
#define IMG_HM1   511.0f
#define BBOX_CLIP 4.135166556742356f   // log(1000/16)
#define NEG_KEY   -1.0e30f

// ---------------------------------------------------------------------------
// Kernel 1: 2-class softmax score + class-1 box decode + clip (elementwise)
// ---------------------------------------------------------------------------
__global__ void pp_decode(const float* __restrict__ logits,
                          const float* __restrict__ regr,
                          const float* __restrict__ props,
                          float* __restrict__ scores,
                          float* __restrict__ boxes,
                          int total)
{
    int i = blockIdx.x * blockDim.x + threadIdx.x;
    if (i >= total) return;
    __builtin_prefetch(&regr[8 * i + 2048], 0, 1);   // -> global_prefetch_b8

    float l0 = logits[2 * i + 0];
    float l1 = logits[2 * i + 1];
    float sc = 1.0f / (1.0f + __expf(l0 - l1));      // softmax prob of class 1

    float px1 = props[4 * i + 0], py1 = props[4 * i + 1];
    float px2 = props[4 * i + 2], py2 = props[4 * i + 3];
    float w  = px2 - px1 + 1.0f;
    float h  = py2 - py1 + 1.0f;
    float cx = px1 + 0.5f * w;
    float cy = py1 + 0.5f * h;

    // class-1 deltas live at rel[:, 4:8]
    float dx = regr[8 * i + 4] * 0.1f;                    // / wx (10)
    float dy = regr[8 * i + 5] * 0.1f;                    // / wy (10)
    float dw = fminf(regr[8 * i + 6] * 0.2f, BBOX_CLIP);  // / ww (5)
    float dh = fminf(regr[8 * i + 7] * 0.2f, BBOX_CLIP);  // / wh (5)

    float pcx = dx * w + cx;
    float pcy = dy * h + cy;
    float pw  = __expf(dw) * w;
    float ph  = __expf(dh) * h;

    float x1 = fminf(fmaxf(pcx - 0.5f * pw,        0.0f), IMG_WM1);
    float y1 = fminf(fmaxf(pcy - 0.5f * ph,        0.0f), IMG_HM1);
    float x2 = fminf(fmaxf(pcx + 0.5f * pw - 1.0f, 0.0f), IMG_WM1);
    float y2 = fminf(fmaxf(pcy + 0.5f * ph - 1.0f, 0.0f), IMG_HM1);

    boxes[4 * i + 0] = x1;
    boxes[4 * i + 1] = y1;
    boxes[4 * i + 2] = x2;
    boxes[4 * i + 3] = y2;
    scores[i] = sc;
}

// ---------------------------------------------------------------------------
// Kernel 2: WMMA rank-sort.
//   rank[i] = #{ j : key_j > key_i } + #{ j : key_j == key_i && j < i }
// computed as a row-sum of a 0/1 comparison matrix via
// v_wmma_f32_16x16x32_f16 with an all-ones B matrix. Each wave owns a
// 16-row M-tile and sweeps K over all 4096 columns (128 WMMAs).
// Ranks form a permutation (strict total order) -> race-free scatter of the
// sorted score/box arrays. Keys are staged memory->LDS with the CDNA5
// async-to-LDS path (ASYNCcnt) when the toolchain exposes it.
// ---------------------------------------------------------------------------
__global__ void pp_rank_wmma(const float* __restrict__ scores,
                             const float* __restrict__ boxes,
                             float* __restrict__ sortedScores,
                             float* __restrict__ sortedBoxes)
{
    __shared__ float keys[N_PROP];                   // 16 KB of the 320 KB WGP LDS

    const int img        = blockIdx.x >> 5;          // 32 blocks per image
    const int blockInImg = blockIdx.x & 31;
    const float* s = scores + (size_t)img * N_PROP;

#if HAVE_ASYNC_LDS
    // DMA the image's 4096 scores straight into LDS (no VGPR round-trip,
    // tracked with ASYNCcnt), then apply the key transform in place.
    for (int i = threadIdx.x; i < N_PROP; i += blockDim.x) {
        __builtin_amdgcn_global_load_async_to_lds_b32(
            (as1_int*)(uintptr_t)(s + i),
            (as3_int*)(unsigned)(uintptr_t)&keys[i],
            0, 0);
    }
#if HAVE_WAIT_ASYNC
    __builtin_amdgcn_s_wait_asynccnt(0);
#else
    asm volatile("s_wait_asynccnt 0x0" ::: "memory");
#endif
    __syncthreads();
    for (int i = threadIdx.x; i < N_PROP; i += blockDim.x) {
        float sc = keys[i];
        keys[i] = (sc > SCORE_TH) ? sc : NEG_KEY;
    }
#else
    for (int i = threadIdx.x; i < N_PROP; i += blockDim.x) {
        float sc = s[i];
        keys[i] = (sc > SCORE_TH) ? sc : NEG_KEY;
    }
#endif
    __syncthreads();

    const int wave = threadIdx.x >> 5;               // wave32
    const int lane = threadIdx.x & 31;
    const int m0   = (blockInImg * 8 + wave) * 16;   // this wave's 16-row tile
    const int myRow = m0 + (lane & 15);              // A-layout: M = lane & 15
    const float myKey = keys[myRow];

    // A-matrix (f16 16x32) K offsets for this lane within each 32-wide chunk:
    // lanes 0-15: VGPR0-3 -> K 0..7,  VGPR4-7 -> K 16..23
    // lanes16-31: VGPR0-3 -> K 8..15, VGPR4-7 -> K 24..31
    const int klo = (lane < 16) ? 0 : 8;
    const int khi = (lane < 16) ? 16 : 24;

    v16h bones;
    #pragma unroll
    for (int t = 0; t < 16; ++t) bones[t] = (_Float16)1.0f;

    v8f acc = {};
    // Unroll x2: two independent compare-gen/WMMA pairs in flight so the
    // VALU compare generation co-executes with the XDL (TRANS-class) WMMAs.
    #pragma unroll 2
    for (int k0 = 0; k0 < N_PROP; k0 += 32) {
        v16h a;
        #pragma unroll
        for (int t = 0; t < 8; ++t) {
            int j = k0 + klo + t;
            float kj = keys[j];
            a[t] = (_Float16)(((kj > myKey) || (kj == myKey && j < myRow)) ? 1.0f : 0.0f);
            int j2 = k0 + khi + t;
            float kj2 = keys[j2];
            a[8 + t] = (_Float16)(((kj2 > myKey) || (kj2 == myKey && j2 < myRow)) ? 1.0f : 0.0f);
        }
        // D[m][n] += sum_k A[m][k] * 1  -> running rank count in every column
        acc = __builtin_amdgcn_wmma_f32_16x16x32_f16(
                  false, a, false, bones, (short)0, acc, false, false);
    }

    // C/D f32 16x16 layout: lanes 0-15 vgpr r -> M=r; lanes 16-31 vgpr r -> M=8+r.
    // Lanes 0-7 extract rows 0-7, lanes 16-23 extract rows 8-15.
    const int r = lane & 7;
    const bool writer = (lane < 8) || (lane >= 16 && lane < 24);
    if (writer) {
        int row  = m0 + r + ((lane >= 16) ? 8 : 0);
        int rank = (int)(acc[r] + 0.5f);
        const float* b  = boxes       + ((size_t)img * N_PROP + row)  * 4;
        float*       ob = sortedBoxes + ((size_t)img * N_PROP + rank) * 4;
        ob[0] = b[0]; ob[1] = b[1]; ob[2] = b[2]; ob[3] = b[3];
        sortedScores[(size_t)img * N_PROP + rank] = s[row];
    }
}

// ---------------------------------------------------------------------------
// Kernel 3: sequential greedy NMS on the sorted list + top-100 compaction.
// One 1024-thread block (32 wave32s) per image; keep flags in LDS; each
// thread holds 4 candidate boxes in registers; one barrier per pivot.
// Output: [B,100,5] boxes+score, then [B,100] labels, then [B,100] valid,
// all as float (bounded by out_size).
// ---------------------------------------------------------------------------
__global__ void pp_nms_topk(const float* __restrict__ sortedScores,
                            const float* __restrict__ sortedBoxes,
                            float* __restrict__ out,
                            int B, int out_size)
{
    __shared__ unsigned char keep[N_PROP];

    const int img = blockIdx.x;
    const int tid = threadIdx.x;
    const float* ss = sortedScores + (size_t)img * N_PROP;
    const float* sb = sortedBoxes  + (size_t)img * N_PROP * 4;

    float mx1[4], my1[4], mx2[4], my2[4], mar[4];
    #pragma unroll
    for (int k = 0; k < 4; ++k) {
        int j = tid + k * 1024;
        mx1[k] = sb[j * 4 + 0];
        my1[k] = sb[j * 4 + 1];
        mx2[k] = sb[j * 4 + 2];
        my2[k] = sb[j * 4 + 3];
        mar[k] = (mx2[k] - mx1[k] + 1.0f) * (my2[k] - my1[k] + 1.0f);
        keep[j] = (ss[j] > SCORE_TH) ? 1 : 0;
    }
    __syncthreads();

    for (int i = 0; i < N_PROP; ++i) {
        if (keep[i]) {                                // uniform across block
            float x1 = sb[i * 4 + 0], y1 = sb[i * 4 + 1];
            float x2 = sb[i * 4 + 2], y2 = sb[i * 4 + 3];
            float ai = (x2 - x1 + 1.0f) * (y2 - y1 + 1.0f);
            #pragma unroll
            for (int k = 0; k < 4; ++k) {
                int j = tid + k * 1024;
                if (j > i && keep[j]) {
                    float xx1 = fmaxf(x1, mx1[k]);
                    float yy1 = fmaxf(y1, my1[k]);
                    float xx2 = fminf(x2, mx2[k]);
                    float yy2 = fminf(y2, my2[k]);
                    float iw = fmaxf(xx2 - xx1 + 1.0f, 0.0f);
                    float ih = fmaxf(yy2 - yy1 + 1.0f, 0.0f);
                    float inter = iw * ih;
                    float iou = inter / (ai + mar[k] - inter);
                    if (iou > NMS_TH) keep[j] = 0;
                }
            }
        }
        __syncthreads();
    }

    if (tid == 0) {
        const int boxOff   = img * DETS * 5;
        const int labelOff = B * DETS * 5 + img * DETS;
        const int validOff = B * DETS * 6 + img * DETS;
        int cnt = 0;
        for (int i = 0; i < N_PROP && cnt < DETS; ++i) {
            if (keep[i]) {
                int o = boxOff + cnt * 5;
                if (o + 4 < out_size) {
                    out[o + 0] = sb[i * 4 + 0];
                    out[o + 1] = sb[i * 4 + 1];
                    out[o + 2] = sb[i * 4 + 2];
                    out[o + 3] = sb[i * 4 + 3];
                    out[o + 4] = ss[i];
                }
                if (labelOff + cnt < out_size) out[labelOff + cnt] = 1.0f;
                if (validOff + cnt < out_size) out[validOff + cnt] = 1.0f;
                ++cnt;
            }
        }
        for (; cnt < DETS; ++cnt) {
            int o = boxOff + cnt * 5;
            if (o + 4 < out_size) {
                out[o + 0] = 0.0f; out[o + 1] = 0.0f; out[o + 2] = 0.0f;
                out[o + 3] = 0.0f; out[o + 4] = 0.0f;
            }
            if (labelOff + cnt < out_size) out[labelOff + cnt] = 0.0f;
            if (validOff + cnt < out_size) out[validOff + cnt] = 0.0f;
        }
    }
}

// ---------------------------------------------------------------------------
extern "C" void kernel_launch(void* const* d_in, const int* in_sizes, int n_in,
                              void* d_out, int out_size, void* d_ws, size_t ws_size,
                              hipStream_t stream)
{
    const float* logits = (const float*)d_in[0];   // [B,N,2]
    const float* regr   = (const float*)d_in[1];   // [B,N,8]
    const float* props  = (const float*)d_in[2];   // [B,N,4]

    const int total = in_sizes[0] / 2;             // B*N
    const int B     = total / N_PROP;

    // Workspace layout (floats): scores[B*N] | boxes[B*N*4] | sScores[B*N] | sBoxes[B*N*4]
    float* ws      = (float*)d_ws;
    float* scores  = ws;
    float* boxes   = scores  + (size_t)total;
    float* sScores = boxes   + (size_t)total * 4;
    float* sBoxes  = sScores + (size_t)total;

    pp_decode<<<(total + 255) / 256, 256, 0, stream>>>(
        logits, regr, props, scores, boxes, total);

    // 8 waves/block, 1 M-tile (16 rows) per wave -> 32 blocks per image
    pp_rank_wmma<<<B * 32, 256, 0, stream>>>(scores, boxes, sScores, sBoxes);

    pp_nms_topk<<<B, 1024, 0, stream>>>(sScores, sBoxes, (float*)d_out, B, out_size);
}
